// CenterTargetMamba_6579889897504
// MI455X (gfx1250) — matrix operand order repaired and checked by reading, not compiled, not run on hardware
//
#include <hip/hip_runtime.h>
#include <hip/hip_bf16.h>
#include <math.h>

// ---------------- model constants ----------------
namespace {
constexpr int BATCH = 512;
constexpr int BANDS = 200;
constexpr int P     = 15;
constexpr int HWN   = P * P;          // 225
constexpr int NCLS  = 16;
constexpr int D0 = 96, D1 = 128, D2 = 160;
constexpr int L0 = 225, L1 = 49, L2 = 9;
constexpr float DEC_MIN = 0.9f, DEC_MAX = 0.999f;
}

typedef __attribute__((ext_vector_type(16))) __bf16          v16bf;
typedef __attribute__((ext_vector_type(16))) unsigned short  v16us;
typedef __attribute__((ext_vector_type(8)))  unsigned short  v8us;
typedef __attribute__((ext_vector_type(4)))  unsigned short  v4us;
typedef __attribute__((ext_vector_type(4)))  float           v4f;
typedef __attribute__((ext_vector_type(8)))  float           v8f;

#define DEV __device__ __forceinline__

DEV unsigned short f2bf(float f) {
  __bf16 h = (__bf16)f;                       // native RNE convert (v_cvt_pk_bf16_f32)
  return __builtin_bit_cast(unsigned short, h);
}
DEV float silu_f(float x) { return x / (1.0f + __expf(-x)); }
DEV float gelu_f(float x) {  // jax.nn.gelu default (tanh approximation)
  const float c = 0.7978845608028654f;
  return 0.5f * x * (1.0f + tanhf(c * (x + 0.044715f * x * x * x)));
}

// ---------------- WMMA GEMM ----------------
// out(MxN) = [res +] act(A(MxK) @ W(KxN) + bias)
// 128 threads (4 waves), 64x64 tile, BK=32, bf16 operands, f32 accumulate.
// Double-buffered LDS, pipelined staging, B stored transposed in LDS so
// fragment builds are pure ds_load_b128. For ABF16 (A already bf16 in global,
// K%32==0, M%64==0 in this model) the A tile is staged with the CDNA5 async
// DMA path: global_load_async_to_lds_b64 + s_wait_asynccnt.
template <bool ABF16, int ACT, bool RES, bool HOUT>
__global__ __launch_bounds__(128) void gemm_wmma_kernel(
    const void* __restrict__ Av, const float* __restrict__ W,
    const float* __restrict__ bias, const float* __restrict__ res,
    float* __restrict__ outF, unsigned short* __restrict__ outH,
    int M, int N, int K) {
  constexpr int LDA = 40;   // halfs per A row (32 + pad, keeps 16B align)
  constexpr int LDB = 40;   // halfs per B column (32 + pad)
  __shared__ unsigned short As[2][64 * LDA];   // [row][k]
  __shared__ unsigned short Bs[2][64 * LDB];   // [col][k]  (transposed)

  const int tid  = threadIdx.x;
  const int lane = tid & 31;
  const int wave = tid >> 5;
  const int tm   = blockIdx.y * 64;
  const int tn   = blockIdx.x * 64;
  const int r16  = lane & 15;   // A row-in-frag / B,D col-in-frag
  const int hi   = lane >> 4;

  const float*          Af = (const float*)Av;
  const unsigned short* Ah = (const unsigned short*)Av;

  v8f acc[4];
#pragma unroll
  for (int j = 0; j < 4; ++j)
#pragma unroll
    for (int e = 0; e < 8; ++e) acc[j][e] = 0.0f;

  // staging registers (tile t+1)
  v4f  aF[4];
  v4f  bF[4];
  const int br0 = (tid >> 4) * 4;   // B k-row group
  const int bc0 = (tid & 15) * 4;   // B col group

  auto loadT = [&](int kt, int buf) {
    const int kb = kt * 32;
    if constexpr (ABF16) {
      // async DMA: A tile bytes go straight to LDS (no VGPR staging).
      unsigned short* Ab = As[buf];
#pragma unroll
      for (int i2 = 0; i2 < 4; ++i2) {
        int gidx = tid + 128 * i2;
        int r = gidx >> 3, c4 = (gidx & 7) * 4;
        unsigned lds = (unsigned)(size_t)(Ab + r * LDA + c4);
        const unsigned short* gp = Ah + (size_t)(tm + r) * K + (kb + c4);
        asm volatile("global_load_async_to_lds_b64 %0, %1, off"
                     :: "v"(lds), "v"(gp) : "memory");
      }
    } else {
#pragma unroll
      for (int i2 = 0; i2 < 4; ++i2) {          // A: 4 float4 runs along k
        int gidx = tid + 128 * i2;
        int r = gidx >> 3, c4 = (gidx & 7) * 4;
        int gr = tm + r, gc = kb + c4;
        v4f t = {0.f, 0.f, 0.f, 0.f};
        if (gr < M && gc < K) t = *(const v4f*)(Af + (size_t)gr * K + gc);
        aF[i2] = t;
      }
    }
#pragma unroll
    for (int j2 = 0; j2 < 4; ++j2) {            // B: 4x4 subtile (rows along k)
      int gk = kb + br0 + j2, gc = tn + bc0;
      v4f t = {0.f, 0.f, 0.f, 0.f};
      if (gk < K && gc < N) t = *(const v4f*)(W + (size_t)gk * N + gc);
      bF[j2] = t;
    }
  };
  auto storeT = [&](int buf) {
    unsigned short* Bb = Bs[buf];
    if constexpr (!ABF16) {
      unsigned short* Ab = As[buf];
#pragma unroll
      for (int i2 = 0; i2 < 4; ++i2) {
        int gidx = tid + 128 * i2;
        int r = gidx >> 3, c4 = (gidx & 7) * 4;
        v4us pk;
        pk[0] = f2bf(aF[i2][0]); pk[1] = f2bf(aF[i2][1]);
        pk[2] = f2bf(aF[i2][2]); pk[3] = f2bf(aF[i2][3]);
        *(v4us*)(Ab + r * LDA + c4) = pk;        // ds_store_b64
      }
    }
#pragma unroll
    for (int i2 = 0; i2 < 4; ++i2) {             // in-register 4x4 transpose
      v4us pk;
      pk[0] = f2bf(bF[0][i2]); pk[1] = f2bf(bF[1][i2]);
      pk[2] = f2bf(bF[2][i2]); pk[3] = f2bf(bF[3][i2]);
      *(v4us*)(Bb + (bc0 + i2) * LDB + br0) = pk; // ds_store_b64 (transposed)
    }
  };

  const int ntiles = (K + 31) >> 5;
  loadT(0, 0);
  storeT(0);
  if constexpr (ABF16) asm volatile("s_wait_asynccnt 0x0" ::: "memory");
  __syncthreads();

  for (int t = 0; t < ntiles; ++t) {
    if (t + 1 < ntiles) loadT(t + 1, (t + 1) & 1);  // next tile (async / regs)

    const unsigned short* Ab = As[t & 1];
    const unsigned short* Bb = Bs[t & 1];
    // A fragment: lane r16 holds K=[16hi..16hi+16) (matches B)
    const unsigned short* ap = Ab + (wave * 16 + r16) * LDA;
    v8us alo = *(const v8us*)(ap + 16 * hi);          // ds_load_b128
    v8us ahi = *(const v8us*)(ap + 16 * hi + 8);     // ds_load_b128
    v16us au = __builtin_shufflevector(alo, ahi, 0, 1, 2, 3, 4, 5, 6, 7,
                                       8, 9, 10, 11, 12, 13, 14, 15);
    v16bf a = __builtin_bit_cast(v16bf, au);
    // hoist all four B fragments: one dscnt wait, then 4 back-to-back WMMAs
    v16us bu[4];
#pragma unroll
    for (int j = 0; j < 4; ++j) {
      const unsigned short* bp = Bb + (j * 16 + r16) * LDB + 16 * hi;
      v8us blo = *(const v8us*)bp;                   // ds_load_b128
      v8us bhi = *(const v8us*)(bp + 8);             // ds_load_b128
      bu[j] = __builtin_shufflevector(blo, bhi, 0, 1, 2, 3, 4, 5, 6, 7,
                                      8, 9, 10, 11, 12, 13, 14, 15);
    }
#pragma unroll
    for (int j = 0; j < 4; ++j)
      acc[j] = __builtin_amdgcn_wmma_f32_16x16x32_bf16(
          false, a, false, __builtin_bit_cast(v16bf, bu[j]), (short)0,
          acc[j], false, false);

#if __has_builtin(__builtin_amdgcn_sched_group_barrier)
    // Pin the schedule: issue all 10 fragment ds_loads first, then the 4
    // WMMAs back-to-back, so dscnt waits are partial instead of a full
    // drain before every matrix op. (0x100 = DS-read, 0x8 = MFMA/WMMA.)
    __builtin_amdgcn_sched_group_barrier(0x100, 10, 0);
    __builtin_amdgcn_sched_group_barrier(0x008, 4, 0);
#endif

    if (t + 1 < ntiles) storeT((t + 1) & 1);     // fill other buffer (B [+A])
    if constexpr (ABF16) asm volatile("s_wait_asynccnt 0x0" ::: "memory");
    __syncthreads();
  }

  // epilogue: D layout — vgpr v: M = v + 8*hi, N = lane&15
#pragma unroll
  for (int j = 0; j < 4; ++j) {
    int gcol = tn + j * 16 + r16;
    if (gcol >= N) continue;
    float bv = bias ? bias[gcol] : 0.0f;
#pragma unroll
    for (int v = 0; v < 8; ++v) {
      int grow = tm + wave * 16 + v + 8 * hi;
      if (grow >= M) continue;
      float val = acc[j][v] + bv;
      if (ACT == 1) val = gelu_f(val);
      if (RES) val += res[(size_t)grow * N + gcol];
      if (HOUT) outH[(size_t)grow * N + gcol] = f2bf(val);
      else      outF[(size_t)grow * N + gcol] = val;
    }
  }
}

// ---------------- LayerNorm: one wave32 per row ----------------
__global__ __launch_bounds__(128) void ln_kernel(
    const float* __restrict__ x, const float* __restrict__ g,
    const float* __restrict__ b, float* __restrict__ y, int rows, int d) {
  int row  = blockIdx.x * 4 + (threadIdx.x >> 5);
  int lane = threadIdx.x & 31;
  if (row >= rows) return;
  const float* xr = x + (size_t)row * d;
  float s = 0.0f;
  for (int c = lane; c < d; c += 32) s += xr[c];
  for (int o = 16; o; o >>= 1) s += __shfl_xor(s, o, 32);
  float mean = s / d;
  float vv = 0.0f;
  for (int c = lane; c < d; c += 32) { float t = xr[c] - mean; vv += t * t; }
  for (int o = 16; o; o >>= 1) vv += __shfl_xor(vv, o, 32);
  float rstd = rsqrtf(vv / d + 1e-5f);
  float* yr = y + (size_t)row * d;
  for (int c = lane; c < d; c += 32) yr[c] = (xr[c] - mean) * rstd * g[c] + b[c];
}

// ---------------- SSM scan: h[t]=a*h[t-1]+silu(u[t]); out=h*silu(z) ----------
__global__ void scan_kernel(const float* __restrict__ xz, float* __restrict__ go,
                            int Bn, int L, int d) {
  int idx = blockIdx.x * blockDim.x + threadIdx.x;
  if (idx >= Bn * d) return;
  int b = idx / d, c = idx % d;
  float a = DEC_MIN + (DEC_MAX - DEC_MIN) * ((d > 1) ? (float)c / (float)(d - 1) : 0.0f);
  const float* u = xz + (size_t)b * L * 2 * d + c;
  const float* z = u + d;
  float* o = go + (size_t)b * L * d + c;
  float h = 0.0f;
  for (int t = 0; t < L; ++t) {
    h = a * h + silu_f(u[(size_t)t * 2 * d]);
    o[(size_t)t * d] = h * silu_f(z[(size_t)t * 2 * d]);
  }
}

// ---------------- gathers ----------------
__global__ void stem_gather(const float* __restrict__ x, float* __restrict__ A) {
  // x:(B,BANDS,225) -> A:(B*225, BANDS)
  size_t i = (size_t)blockIdx.x * blockDim.x + threadIdx.x;
  size_t total = (size_t)BATCH * HWN * BANDS;
  if (i >= total) return;
  int c = (int)(i % BANDS);
  size_t r = i / BANDS;
  int hw = (int)(r % HWN);
  int b  = (int)(r / HWN);
  A[i] = x[((size_t)b * BANDS + c) * HWN + hw];
}

__global__ void down_gather(const float* __restrict__ X, float* __restrict__ A,
                            int Bn, int h, int w, int C, int nh, int nw) {
  // A:(Bn*nh*nw, 4C), col = (ky*2+kx)*C + c  (matches HWIO weight flatten)
  size_t i = (size_t)blockIdx.x * blockDim.x + threadIdx.x;
  size_t total = (size_t)Bn * nh * nw * 4 * C;
  if (i >= total) return;
  int c = (int)(i % C);
  int k = (int)((i / C) % 4);
  size_t r = i / ((size_t)4 * C);
  int ow = (int)(r % nw);
  int oh = (int)((r / nw) % nh);
  int b  = (int)(r / ((size_t)nw * nh));
  int ky = k >> 1, kx = k & 1;
  A[i] = X[((size_t)b * h * w + (size_t)(2 * oh + ky) * w + (2 * ow + kx)) * C + c];
}

// ---------------- pooling / heads ----------------
__global__ void mmean_kernel(const float* __restrict__ Y, float* __restrict__ out,
                             int Bn, int L, int d, unsigned mask) {
  int idx = blockIdx.x * blockDim.x + threadIdx.x;
  if (idx >= Bn * d) return;
  int b = idx / d, c = idx % d;
  float s = 0.0f; float cnt = 0.0f;
  for (int t = 0; t < L; ++t)
    if ((mask >> t) & 1u) { s += Y[((size_t)b * L + t) * d + c]; cnt += 1.0f; }
  out[idx] = s / fmaxf(cnt, 1e-6f);
}

__global__ void take_token(const float* __restrict__ Y, float* __restrict__ out,
                           int Bn, int L, int d, int t) {
  int idx = blockIdx.x * blockDim.x + threadIdx.x;
  if (idx >= Bn * d) return;
  out[idx] = Y[((size_t)(idx / d) * L + t) * d + (idx % d)];
}

__global__ void build3_kernel(const float* __restrict__ p, const float* __restrict__ c,
                              float* __restrict__ out, int Bn, int d) {
  // [pooled, center, pooled*center]
  int idx = blockIdx.x * blockDim.x + threadIdx.x;
  int W = 3 * d;
  if (idx >= Bn * W) return;
  int b = idx / W, col = idx % W;
  float v;
  if (col < d)            v = p[b * d + col];
  else if (col < 2 * d)   v = c[b * d + col - d];
  else { int cc = col - 2 * d; v = p[b * d + cc] * c[b * d + cc]; }
  out[idx] = v;
}

__global__ void build5_kernel(const float* __restrict__ p, const float* __restrict__ c,
                              const float* __restrict__ t1, const float* __restrict__ t2,
                              float* __restrict__ out, int Bn, int d, int mode) {
  // [pooled, center, t1, t2, mode? t1*t2 : t1-t2]
  int idx = blockIdx.x * blockDim.x + threadIdx.x;
  int W = 5 * d;
  if (idx >= Bn * W) return;
  int b = idx / W, col = idx % W;
  float v;
  if (col < d)            v = p[b * d + col];
  else if (col < 2 * d)   v = c[b * d + col - d];
  else if (col < 3 * d)   v = t1[b * d + col - 2 * d];
  else if (col < 4 * d)   v = t2[b * d + col - 3 * d];
  else {
    int cc = col - 4 * d;
    v = mode ? t1[b * d + cc] * t2[b * d + cc] : t1[b * d + cc] - t2[b * d + cc];
  }
  out[idx] = v;
}

__global__ void rsa_stats_kernel(const float* __restrict__ x, float* __restrict__ out) {
  // out:(B, 600) = [center_pixel, mean, std] per band
  int idx = blockIdx.x * blockDim.x + threadIdx.x;
  if (idx >= BATCH * BANDS) return;
  int band = idx % BANDS, b = idx / BANDS;
  const float* pp = x + ((size_t)b * BANDS + band) * HWN;
  float s = 0.0f, s2 = 0.0f;
  for (int i = 0; i < HWN; ++i) { float v = pp[i]; s += v; s2 += v * v; }
  float mean = s / (float)HWN;
  float var  = fmaxf(s2 / (float)HWN - mean * mean, 0.0f);
  out[(size_t)b * 600 + band]             = pp[(P / 2) * P + (P / 2)];
  out[(size_t)b * 600 + BANDS + band]     = mean;
  out[(size_t)b * 600 + 2 * BANDS + band] = sqrtf(var);
}

__global__ void feat_kernel(const float* __restrict__ p, const float* __restrict__ a,
                            const float* __restrict__ b, const float* __restrict__ c,
                            const float* __restrict__ d, float* __restrict__ f, int n) {
  int i = blockIdx.x * blockDim.x + threadIdx.x;
  if (i >= n) return;
  f[i] = p[i] + 0.5f * (a[i] + b[i] + c[i] + d[i]);
}

// ---------------- host-side GEMM dispatcher ----------------
static inline void gemm(hipStream_t s, const void* A, bool abf16, const float* W,
                        const float* bias, const float* res, void* out, bool hout,
                        int act, int M, int N, int K) {
  dim3 grid((N + 63) / 64, (M + 63) / 64), block(128);
  float* oF = hout ? nullptr : (float*)out;
  unsigned short* oH = hout ? (unsigned short*)out : nullptr;
  if (abf16 && res)
    gemm_wmma_kernel<true, 0, true, false><<<grid, block, 0, s>>>(A, W, bias, res, oF, oH, M, N, K);
  else if (act == 1 && hout)
    gemm_wmma_kernel<false, 1, false, true><<<grid, block, 0, s>>>(A, W, bias, res, oF, oH, M, N, K);
  else if (act == 1)
    gemm_wmma_kernel<false, 1, false, false><<<grid, block, 0, s>>>(A, W, bias, res, oF, oH, M, N, K);
  else if (res)
    gemm_wmma_kernel<false, 0, true, false><<<grid, block, 0, s>>>(A, W, bias, res, oF, oH, M, N, K);
  else
    gemm_wmma_kernel<false, 0, false, false><<<grid, block, 0, s>>>(A, W, bias, res, oF, oH, M, N, K);
}

// ---------------- parameter plumbing ----------------
struct StageP {
  const float *ln1_g, *ln1_b, *in_w, *in_b, *out_w, *out_b;
  const float *ln2_g, *ln2_b, *m1_w, *m1_b, *m2_w, *m2_b;
};
struct ParamsP {
  const float *x;
  const float *stem_w, *stem_b, *down01_w, *down01_b, *down12_w, *down12_b;
  const float *lnh_g, *lnh_b;
  const float *cca_w1, *cca_b1, *cca_w2, *cca_b2;
  const float *cpa_w1, *cpa_b1, *cpa_w2, *cpa_b2;
  const float *bca_w1, *bca_b1, *bca_w2, *bca_b2;
  const float *rsa_w1, *rsa_b1, *rsa_w2, *rsa_b2;
  const float *back_w, *back_b, *head_w, *head_b;
  StageP st[3];
};

#define EL(K, total, ...) K<<<((int)(((total) + 255) / 256)), 256, 0, stream>>>(__VA_ARGS__)

extern "C" void kernel_launch(void* const* d_in, const int* in_sizes, int n_in,
                              void* d_out, int out_size, void* d_ws, size_t ws_size,
                              hipStream_t stream) {
  auto g = [&](int i) { return (const float*)d_in[i]; };
  // locate x (uniquely 512*200*225 elements); infer flatten order of params
  int xidx = 0;
  for (int i = 0; i < n_in; ++i)
    if (in_sizes[i] == BATCH * BANDS * HWN) { xidx = i; break; }
  int b0 = (xidx == 0) ? 1 : 0;

  ParamsP pp{};
  pp.x = g(xidx);
  if (in_sizes[b0] == 320) {
    // jax-pytree sorted-key order: params leaves alphabetical
    int i = b0;
    pp.back_b = g(i + 0);  pp.back_w = g(i + 1);
    pp.bca_b1 = g(i + 2);  pp.bca_b2 = g(i + 3);  pp.bca_w1 = g(i + 4);  pp.bca_w2 = g(i + 5);
    pp.cca_b1 = g(i + 6);  pp.cca_b2 = g(i + 7);  pp.cca_w1 = g(i + 8);  pp.cca_w2 = g(i + 9);
    pp.cpa_b1 = g(i + 10); pp.cpa_b2 = g(i + 11); pp.cpa_w1 = g(i + 12); pp.cpa_w2 = g(i + 13);
    pp.down01_b = g(i + 14); pp.down01_w = g(i + 15);
    pp.down12_b = g(i + 16); pp.down12_w = g(i + 17);
    pp.head_b = g(i + 18); pp.head_w = g(i + 19);
    pp.lnh_b  = g(i + 20); pp.lnh_g  = g(i + 21);
    pp.rsa_b1 = g(i + 22); pp.rsa_b2 = g(i + 23); pp.rsa_w1 = g(i + 24); pp.rsa_w2 = g(i + 25);
    int j = i + 26;
    for (int s = 0; s < 3; ++s, j += 12) {
      pp.st[s].in_b = g(j + 0);  pp.st[s].in_w = g(j + 1);
      pp.st[s].ln1_b = g(j + 2); pp.st[s].ln1_g = g(j + 3);
      pp.st[s].ln2_b = g(j + 4); pp.st[s].ln2_g = g(j + 5);
      pp.st[s].m1_b = g(j + 6);  pp.st[s].m1_w = g(j + 7);
      pp.st[s].m2_b = g(j + 8);  pp.st[s].m2_w = g(j + 9);
      pp.st[s].out_b = g(j + 10); pp.st[s].out_w = g(j + 11);
    }
    pp.stem_b = g(j + 0); pp.stem_w = g(j + 1);
  } else {
    // insertion order as written in setup_inputs()
    int j = b0;
    pp.stem_w = g(j); pp.stem_b = g(j + 1); j += 2;
    for (int s = 0; s < 3; ++s) {
      pp.st[s].ln1_g = g(j + 0); pp.st[s].ln1_b = g(j + 1);
      pp.st[s].in_w = g(j + 2);  pp.st[s].in_b = g(j + 3);
      pp.st[s].out_w = g(j + 4); pp.st[s].out_b = g(j + 5);
      pp.st[s].ln2_g = g(j + 6); pp.st[s].ln2_b = g(j + 7);
      pp.st[s].m1_w = g(j + 8);  pp.st[s].m1_b = g(j + 9);
      pp.st[s].m2_w = g(j + 10); pp.st[s].m2_b = g(j + 11);
      j += 12;
      if (s == 0) { pp.down01_w = g(j); pp.down01_b = g(j + 1); j += 2; }
      if (s == 1) { pp.down12_w = g(j); pp.down12_b = g(j + 1); j += 2; }
    }
    pp.lnh_g = g(j); pp.lnh_b = g(j + 1); j += 2;
    pp.cca_w1 = g(j); pp.cca_b1 = g(j + 1); pp.cca_w2 = g(j + 2); pp.cca_b2 = g(j + 3); j += 4;
    pp.cpa_w1 = g(j); pp.cpa_b1 = g(j + 1); pp.cpa_w2 = g(j + 2); pp.cpa_b2 = g(j + 3); j += 4;
    pp.bca_w1 = g(j); pp.bca_b1 = g(j + 1); pp.bca_w2 = g(j + 2); pp.bca_b2 = g(j + 3); j += 4;
    pp.rsa_w1 = g(j); pp.rsa_b1 = g(j + 1); pp.rsa_w2 = g(j + 2); pp.rsa_b2 = g(j + 3); j += 4;
    pp.back_w = g(j); pp.back_b = g(j + 1); j += 2;
    pp.head_w = g(j); pp.head_b = g(j + 1);
  }

  // ---------------- workspace layout (floats) ----------------
  float* ws = (float*)d_ws;
  const size_t TOK = (size_t)BATCH * L0 * D0;       // 11,059,200 (max token buf)
  const size_t XZC = (size_t)BATCH * HWN * BANDS;   // 23,040,000 (stem im2col / max)
  float* X  = ws;                 // current tokens (fp32)
  float* Yb = X + TOK;            // LN output / gated scan output (fp32)
  float* XZ = Yb + TOK;           // xz GEMM out (fp32) / MLP hidden (bf16) / im2col
  float* HD = XZ + XZC;           // head scratch
  const int BD2 = BATCH * D2;
  float* pooled  = HD;             float* center  = pooled + BD2;
  float* boundary= center + BD2;
  float* ccab = boundary + BD2;    float* cpab = ccab + BD2;
  float* bcab = cpab + BD2;        float* rsab = bcab + BD2;
  float* feat = rsab + BD2;
  float* cca_in = feat + BD2;                 // 512*480
  float* cpa_in = cca_in + (size_t)BATCH * 480;   // 512*800
  float* bca_in = cpa_in + (size_t)BATCH * 800;   // 512*800
  float* rsa_in = bca_in + (size_t)BATCH * 800;   // 512*600
  float* T1     = rsa_in + (size_t)BATCH * 600;   // 512*320 max hidden

  // ---------------- stem: tok = x^T @ stem_w + stem_b ----------------
  EL(stem_gather, (size_t)BATCH * HWN * BANDS, pp.x, XZ);
  gemm(stream, XZ, false, pp.stem_w, pp.stem_b, nullptr, X, false, 0,
       BATCH * HWN, D0, BANDS);

  // ---------------- stages ----------------
  const int sd[3] = {D0, D1, D2};
  const int sl[3] = {L0, L1, L2};
  const int sn[3] = {2, 2, 5};
  int curH = P, curW = P;
  for (int s = 0; s < 3; ++s) {
    const int d = sd[s], L = sl[s], nb = sn[s];
    const int rows = BATCH * L;
    const StageP& st = pp.st[s];
    for (int k = 0; k < nb; ++k) {
      const float *ln1g = st.ln1_g + (size_t)k * d,  *ln1b = st.ln1_b + (size_t)k * d;
      const float *inw  = st.in_w  + (size_t)k * d * 2 * d, *inb = st.in_b + (size_t)k * 2 * d;
      const float *outw = st.out_w + (size_t)k * d * d, *outb = st.out_b + (size_t)k * d;
      const float *ln2g = st.ln2_g + (size_t)k * d,  *ln2b = st.ln2_b + (size_t)k * d;
      const float *m1w  = st.m1_w  + (size_t)k * d * 4 * d, *m1b = st.m1_b + (size_t)k * 4 * d;
      const float *m2w  = st.m2_w  + (size_t)k * 4 * d * d, *m2b = st.m2_b + (size_t)k * d;

      ln_kernel<<<(rows + 3) / 4, 128, 0, stream>>>(X, ln1g, ln1b, Yb, rows, d);
      gemm(stream, Yb, false, inw, inb, nullptr, XZ, false, 0, rows, 2 * d, d);
      EL(scan_kernel, BATCH * d, XZ, Yb, BATCH, L, d);                 // G -> Yb
      gemm(stream, Yb, false, outw, outb, X, X, false, 0, rows, d, d); // X += G@W+b
      ln_kernel<<<(rows + 3) / 4, 128, 0, stream>>>(X, ln2g, ln2b, Yb, rows, d);
      gemm(stream, Yb, false, m1w, m1b, nullptr, XZ, true, 1, rows, 4 * d, d); // bf16 hidden
      gemm(stream, XZ, true, m2w, m2b, X, X, false, 0, rows, d, 4 * d);        // X += MLP
    }
    if (s < 2) {
      int nh = (curH - 2) / 2 + 1, nw = (curW - 2) / 2 + 1;
      const float* dw = (s == 0) ? pp.down01_w : pp.down12_w;
      const float* db = (s == 0) ? pp.down01_b : pp.down12_b;
      int Cn = sd[s + 1];
      EL(down_gather, (size_t)BATCH * nh * nw * 4 * d, X, XZ, BATCH, curH, curW, d, nh, nw);
      gemm(stream, XZ, false, dw, db, nullptr, X, false, 0, BATCH * nh * nw, Cn, 4 * d);
      curH = nh; curW = nw;
    }
  }

  // ---------------- heads (3x3 token grid) ----------------
  const int rows2 = BATCH * L2;
  ln_kernel<<<(rows2 + 3) / 4, 128, 0, stream>>>(X, pp.lnh_g, pp.lnh_b, Yb, rows2, D2);
  // pooled over all 9; 3x3 cheb masks r=1,r=2 are all-ones => inner=outer=pooled;
  // norm_radius: core(d<=0.35)=center only; boundary(d>=0.7)=8 ring cells.
  EL(mmean_kernel, BD2, Yb, pooled, BATCH, L2, D2, 0x1FFu);
  EL(take_token,   BD2, Yb, center, BATCH, L2, D2, 4);
  EL(mmean_kernel, BD2, Yb, boundary, BATCH, L2, D2, 0x1EFu);

  EL(build3_kernel, BATCH * 3 * D2, pooled, center, cca_in, BATCH, D2);
  gemm(stream, cca_in, false, pp.cca_w1, pp.cca_b1, nullptr, T1, false, 1, BATCH, D2, 3 * D2);
  gemm(stream, T1, false, pp.cca_w2, pp.cca_b2, nullptr, ccab, false, 0, BATCH, D2, D2);

  EL(build5_kernel, BATCH * 5 * D2, pooled, center, pooled, pooled, cpa_in, BATCH, D2, 0);
  gemm(stream, cpa_in, false, pp.cpa_w1, pp.cpa_b1, nullptr, T1, false, 1, BATCH, D2, 5 * D2);
  gemm(stream, T1, false, pp.cpa_w2, pp.cpa_b2, nullptr, cpab, false, 0, BATCH, D2, D2);

  EL(build5_kernel, BATCH * 5 * D2, pooled, center, center, boundary, bca_in, BATCH, D2, 1);
  gemm(stream, bca_in, false, pp.bca_w1, pp.bca_b1, nullptr, T1, false, 1, BATCH, D2, 5 * D2);
  gemm(stream, T1, false, pp.bca_w2, pp.bca_b2, nullptr, bcab, false, 0, BATCH, D2, D2);

  EL(rsa_stats_kernel, BATCH * BANDS, pp.x, rsa_in);
  gemm(stream, rsa_in, false, pp.rsa_w1, pp.rsa_b1, nullptr, T1, false, 1, BATCH, 2 * D2, 3 * BANDS);
  gemm(stream, T1, false, pp.rsa_w2, pp.rsa_b2, nullptr, rsab, false, 0, BATCH, D2, 2 * D2);

  EL(feat_kernel, BD2, pooled, ccab, cpab, bcab, rsab, feat, BD2);
  gemm(stream, feat, false, pp.back_w, pp.back_b, nullptr, T1, false, 1, BATCH, 2 * D2, D2);
  gemm(stream, T1, false, pp.head_w, pp.head_b, nullptr, (float*)d_out, false, 0,
       BATCH, NCLS, 2 * D2);
  (void)out_size; (void)ws_size; (void)n_in;
}
